// GATLayer_20770461843840
// MI455X (gfx1250) — matrix operand order repaired and checked by reading, not compiled, not run on hardware
//
#include <hip/hip_runtime.h>
#include <math.h>

typedef __attribute__((ext_vector_type(16))) _Float16 v16h;
typedef __attribute__((ext_vector_type(8)))  float    v8f;

#define D_IN  256
#define HF    256
#define HEADS 8
#define NCT   33   // 16 proj col-tiles + 16 skip col-tiles + 1 scores tile

// ---- gfx1250 async global->LDS copy (32B per thread), tracked by ASYNCcnt ----
// INST_OFFSET is added to BOTH the global and LDS addresses (ISA 08_async_tensor §4.4),
// so one (lds, global) address pair + offset:16 moves 32 contiguous bytes.
__device__ __forceinline__ void async_copy32(const _Float16* g, _Float16* l) {
  unsigned ldsa = (unsigned)(size_t)l;  // flat->LDS: low 32 bits are the DS offset
  asm volatile(
      "global_load_async_to_lds_b128 %0, %1, off\n\t"
      "global_load_async_to_lds_b128 %0, %1, off offset:16"
      :: "v"(ldsa), "v"(g) : "memory");
}
__device__ __forceinline__ void async_wait0() {
  asm volatile("s_wait_asynccnt 0x0" ::: "memory");
}

// ---------- setup: Wc[256][16] = W_proj @ [W_src | W_tgt] ----------
__global__ void k_prep_wc(const float* __restrict__ Wp, const float* __restrict__ Ws,
                          const float* __restrict__ Wt, float* __restrict__ Wc) {
  int k = blockIdx.x * blockDim.x + threadIdx.x;
  if (k >= D_IN) return;
  for (int j = 0; j < 16; ++j) {
    const float* w = (j < 8) ? Ws : Wt;
    int jj = j & 7;
    float acc = 0.f;
    for (int i = 0; i < HF; ++i) acc += Wp[k * HF + i] * w[i * HEADS + jj];
    Wc[k * 16 + j] = acc;
  }
}

// ---------- pack weights into WMMA B-fragment layout (f16) ----------
// fragment (ct,kt): lane l holds column (l&15), K = kt*32 + e + 16*(l>>4), e=0..15
__global__ void k_pack(const float* __restrict__ Wp, const float* __restrict__ Wk,
                       const float* __restrict__ Wc, _Float16* __restrict__ Bp) {
  int idx = blockIdx.x * blockDim.x + threadIdx.x;
  if (idx >= NCT * 8 * 512) return;
  int frag = idx >> 9;            // ct*8 + kt
  int r    = idx & 511;
  int lane = r >> 4;
  int e    = r & 15;
  int ct = frag >> 3, kt = frag & 7;
  int K   = kt * 32 + e + 16 * (lane >> 4);
  int col = lane & 15;
  float v;
  if (ct < 16)      v = Wp[K * HF + ct * 16 + col];
  else if (ct < 32) v = Wk[K * HF + (ct - 16) * 16 + col];
  else              v = Wc[K * 16 + col];
  Bp[idx] = (_Float16)v;
}

__global__ void k_zero(float* __restrict__ p, int n) {
  int i = blockIdx.x * blockDim.x + threadIdx.x;
  if (i < n) p[i] = 0.f;
}

// ---------- fused GEMM: proj / skip(->accum) / scores from one A pass ----------
// Block = 8 waves = 8 row-tiles (128 rows). B fragments staged through LDS via
// async global->LDS copies, double-buffered, shared by all 8 waves.
__global__ void __launch_bounds__(256)
k_gemm(const float* __restrict__ X, const _Float16* __restrict__ Bp,
       const float* __restrict__ bsrc, const float* __restrict__ btgt,
       float* __restrict__ proj, float* __restrict__ accum,
       float* __restrict__ scores, int nTiles) {
  __shared__ __align__(32) _Float16 Bs[2][8 * 512];  // 2 x 8KB

  int tid  = threadIdx.x;
  int wave = tid >> 5;
  int lane = tid & 31;
  int tile = blockIdx.x * 8 + wave;
  bool valid = (tile < nTiles);
  int tcl = valid ? tile : (nTiles - 1);   // clamp: all waves keep hitting barriers

  int mrow  = lane & 15;
  int half_ = lane >> 4;
  const float* xrow = X + (size_t)(tcl * 16 + mrow) * D_IN;

  // stage col-tile 0 asynchronously while we build A fragments
  async_copy32(Bp + tid * 16, &Bs[0][tid * 16]);

  // A fragments: 16x32 f16, contiguous layout (matches packed B):
  //  e in [0,16): K = kt*32 + 16*half + e
  v16h a[8];
#pragma unroll
  for (int kt = 0; kt < 8; ++kt) {
    int base = kt * 32 + 16 * half_;
#pragma unroll
    for (int e = 0; e < 8; ++e) {
      a[kt][e]     = (_Float16)xrow[base + e];
      a[kt][8 + e] = (_Float16)xrow[base + 8 + e];
    }
  }

  async_wait0();
  __syncthreads();

  int col  = lane & 15;
  int row0 = tcl * 16 + 8 * half_;  // C/D: VGPR r -> row r + 8*half, col = lane&15

  for (int ct = 0; ct < NCT; ++ct) {
    int cur = ct & 1;
    // async-prefetch next col-tile's fragments into the other buffer
    if (ct + 1 < NCT)
      async_copy32(Bp + (size_t)(ct + 1) * 4096 + tid * 16, &Bs[cur ^ 1][tid * 16]);

    // pull the 8 B fragments from LDS (broadcast across waves) + WMMA chain
    v16h b[8];
#pragma unroll
    for (int kt = 0; kt < 8; ++kt)
      b[kt] = *(const v16h*)&Bs[cur][kt * 512 + lane * 16];

    v8f acc = {};
#pragma unroll
    for (int kt = 0; kt < 8; ++kt)
      acc = __builtin_amdgcn_wmma_f32_16x16x32_f16(
          /*neg_a=*/false, a[kt], /*neg_b=*/false, b[kt],
          /*c_mod=*/(short)0, acc, /*reuse_a=*/false, /*reuse_b=*/false);

    if (valid) {
      if (ct < 16) {
#pragma unroll
        for (int r = 0; r < 8; ++r)
          proj[(size_t)(row0 + r) * HF + ct * 16 + col] = acc[r];
      } else if (ct < 32) {
#pragma unroll
        for (int r = 0; r < 8; ++r)
          accum[(size_t)(row0 + r) * HF + (ct - 16) * 16 + col] = acc[r];
      } else {
        float bias = (col < 8) ? bsrc[col] : btgt[col - 8];
#pragma unroll
        for (int r = 0; r < 8; ++r)
          scores[(size_t)(row0 + r) * 16 + col] = acc[r] + bias;
      }
    }
    async_wait0();     // own async stores to Bs[cur^1] complete (overlapped w/ WMMAs)
    __syncthreads();   // publish staging; everyone done reading buf before reuse
  }
}

// ---------- edge pass 1: w = exp(elu(fs+ft)); denom[tgt,h] += w ----------
__global__ void k_edge1(const int* __restrict__ ei, const float* __restrict__ scores,
                        float* __restrict__ ew, float* __restrict__ denom, int E) {
  int t = blockIdx.x * blockDim.x + threadIdx.x;
  if (t >= E * HEADS) return;
  int e = t >> 3, h = t & 7;
  int s = ei[e], d = ei[E + e];
  float f  = scores[(size_t)s * 16 + h] + scores[(size_t)d * 16 + 8 + h];
  float fe = (f > 0.f) ? f : (expf(f) - 1.f);   // elu
  float w  = expf(fe);                          // global-max shift cancels in softmax
  ew[t] = w;
  atomicAdd(&denom[(size_t)d * HEADS + h], w);
}

// ---------- edge pass 2: accum[tgt] += proj[src] * att (wave per edge) ----------
__global__ void __launch_bounds__(256)
k_edge2(const int* __restrict__ ei, const float* __restrict__ proj,
        const float* __restrict__ ew, const float* __restrict__ denom,
        float* __restrict__ accum, int E) {
  int wave = threadIdx.x >> 5, lane = threadIdx.x & 31;
  int e = blockIdx.x * (blockDim.x >> 5) + wave;
  if (e >= E) return;
  int s = ei[e], d = ei[E + e];

  // lanes 0..7 fetch the per-head scalars once; broadcast via shuffles
  float wv = 0.f, dv = 1.f;
  if (lane < HEADS) {
    wv = ew[(size_t)e * HEADS + lane];
    dv = denom[(size_t)d * HEADS + lane];
  }

  const float* ps = proj  + (size_t)s * HF;
  float*       pd = accum + (size_t)d * HF;
#pragma unroll
  for (int h = 0; h < HEADS; ++h) {
    float att = __shfl(wv, h) / (__shfl(dv, h) + 1e-10f);
    atomicAdd(&pd[h * 32 + lane], ps[h * 32 + lane] * att);
  }
}

// ---------- final: out = elu(accum + bias) ----------
__global__ void k_final(const float* __restrict__ accum, const float* __restrict__ bias,
                        float* __restrict__ out, int n4) {
  int i = blockIdx.x * blockDim.x + threadIdx.x;
  if (i >= n4) return;
  float4 v = ((const float4*)accum)[i];
  float4 b = ((const float4*)bias)[i & 63];   // 256 cols = 64 float4
  float4 o;
  o.x = v.x + b.x; o.y = v.y + b.y; o.z = v.z + b.z; o.w = v.w + b.w;
  o.x = (o.x > 0.f) ? o.x : expf(o.x) - 1.f;
  o.y = (o.y > 0.f) ? o.y : expf(o.y) - 1.f;
  o.z = (o.z > 0.f) ? o.z : expf(o.z) - 1.f;
  o.w = (o.w > 0.f) ? o.w : expf(o.w) - 1.f;
  ((float4*)out)[i] = o;
}

extern "C" void kernel_launch(void* const* d_in, const int* in_sizes, int n_in,
                              void* d_out, int out_size, void* d_ws, size_t ws_size,
                              hipStream_t stream) {
  const float* X    = (const float*)d_in[0];
  const int*   ei   = (const int*)  d_in[1];
  const float* Wp   = (const float*)d_in[2];
  const float* Ws   = (const float*)d_in[3];
  const float* bs   = (const float*)d_in[4];
  const float* Wt   = (const float*)d_in[5];
  const float* bt   = (const float*)d_in[6];
  const float* Wk   = (const float*)d_in[7];
  const float* bias = (const float*)d_in[8];

  const int N = in_sizes[0] / D_IN;
  const int E = in_sizes[1] / 2;

  // workspace carve-up (256B aligned)
  char* w = (char*)d_ws;
  size_t off = 0;
  auto take = [&](size_t bytes) -> char* {
    char* p = w + off;
    off = (off + bytes + 255) & ~(size_t)255;
    return p;
  };
  float*    proj   = (float*)   take((size_t)N * HF * 4);
  float*    accum  = (float*)   take((size_t)N * HF * 4);
  float*    scores = (float*)   take((size_t)N * 16 * 4);
  float*    denom  = (float*)   take((size_t)N * HEADS * 4);
  float*    ew     = (float*)   take((size_t)E * HEADS * 4);
  float*    Wc     = (float*)   take((size_t)D_IN * 16 * 4);
  _Float16* Bp     = (_Float16*)take((size_t)NCT * 8 * 512 * 2);
  (void)ws_size; (void)n_in; (void)out_size;

  k_prep_wc<<<1, 256, 0, stream>>>(Wp, Ws, Wt, Wc);
  k_pack<<<(NCT * 8 * 512 + 255) / 256, 256, 0, stream>>>(Wp, Wk, Wc, Bp);
  k_zero<<<(N * HEADS + 255) / 256, 256, 0, stream>>>(denom, N * HEADS);

  int nTiles = N / 16;  // N = 20000 -> 1250
  k_gemm<<<(nTiles + 7) / 8, 256, 0, stream>>>(X, Bp, bs, bt, proj, accum, scores, nTiles);

  k_edge1<<<((size_t)E * HEADS + 255) / 256, 256, 0, stream>>>(ei, scores, ew, denom, E);
  k_edge2<<<(E + 7) / 8, 256, 0, stream>>>(ei, proj, ew, denom, accum, E);

  k_final<<<(N * (HF / 4) + 255) / 256, 256, 0, stream>>>(accum, bias, (float*)d_out, N * (HF / 4));
}
